// MultiHeadAttention_77257871720517
// MI455X (gfx1250) — matrix operand order repaired and checked by reading, not compiled, not run on hardware
//
#include <hip/hip_runtime.h>
#include <hip/hip_bf16.h>

typedef __attribute__((ext_vector_type(16))) _Float16 v16h;
typedef __attribute__((ext_vector_type(8)))  float    v8f;
typedef __fp16 __attribute__((ext_vector_type(2)))    v2fp16;   // return type of cvt_pkrtz

#define BM 128
#define BN 64
#define BK 64
#define LDSW (BK + 4)    // 68 floats = 272B = 17*16B: async 16B chunks stay aligned for every
                         // row; 68*dr mod 64 = 4*dr distinct for dr=0..15 -> conflict-free reads
#define THREADS 256

union F16Frag { v16h h; unsigned u[8]; };

// Pack two f32 into packed f16 (v_cvt_pk_rtz_f16_f32).
__device__ __forceinline__ unsigned pk16(float a, float b) {
    union { v2fp16 h; unsigned u; } w;
    w.h = __builtin_amdgcn_cvt_pkrtz(a, b);
    return w.u;
}

// Generic C[m,n] = scale * sum_k A[m,k] * B[n,k] + bias[n], f32 in memory,
// f16 WMMA with f32 accumulation. Double-buffered LDS; tiles streamed with
// GLOBAL_LOAD_ASYNC_TO_LDS_B128 and pipelined via ASYNCcnt thresholds
// (s_wait_asynccnt <= next-stage-in-flight => current stage complete).
//   BCONTIG: B's K dimension contiguous -> async B staging (else scalar loads)
//   NT_A:    A tiles streamed with th:TH_LOAD_NT (P matrix, 402MB >> L2)
//   NT_C:    C stored nontemporal (logits, streaming 402MB would evict Q/K from L2)
template<bool BCONTIG, bool NT_A, bool NT_C>
__global__ __launch_bounds__(THREADS)
void wmma_gemm_f32(const float* __restrict__ A, long long sAb, long long sAh,
                   long long ldaR,
                   const float* __restrict__ Bm, long long sBb, long long sBh,
                   long long ldbN, long long ldbK,
                   float* __restrict__ C, long long sCb, long long sCh, long long ldc,
                   int Kdim, int Hdim, const float* __restrict__ bias, float scale)
{
    __shared__ float Asm2[2][BM * LDSW];   // 2 x 34 KB
    __shared__ float Bsm2[2][BN * LDSW];   // 2 x 17 KB

    const int z  = blockIdx.z;
    const int bb = z / Hdim, hh = z % Hdim;
    const long long tileM = (long long)blockIdx.x * BM;
    const long long tileN = (long long)blockIdx.y * BN;

    A  += bb * sAb + hh * sAh + tileM * ldaR;        // A is K-contiguous in all uses
    Bm += bb * sBb + hh * sBh + tileN * ldbN;
    C  += bb * sCb + hh * sCh;

    const int tid  = threadIdx.x;
    const int lane = tid & 31;
    const int wave = tid >> 5;
    const int wm   = (wave & 3) * 32;   // 4 waves down M
    const int wn   = (wave >> 2) * 32;  // 2 waves across N
    const int ml   = lane & 15;
    const int hf   = lane >> 4;         // wave32 half-select per ISA 7.12.2

    // Issue one stage's tile copies into LDS buffer `buf` for K offset k0.
    auto issue_stage = [&](int k0, int buf) {
        // A tile: 128x64 f32 = 2048 async 16B chunks -> 8 wave-instructions
        #pragma unroll
        for (int i = 0; i < 8; ++i) {
            int chunk = i * THREADS + tid;
            int r  = chunk >> 4;            // 16 chunks per 64-float row
            int c4 = (chunk & 15) << 2;
            unsigned lds = (unsigned)(uintptr_t)(&Asm2[buf][r * LDSW + c4]);
            unsigned off = (unsigned)((r * (int)ldaR + k0 + c4) * 4);
            if constexpr (NT_A)
                asm volatile("global_load_async_to_lds_b128 %0, %1, %2 th:TH_LOAD_NT"
                             :: "v"(lds), "v"(off), "s"(A) : "memory");
            else
                asm volatile("global_load_async_to_lds_b128 %0, %1, %2"
                             :: "v"(lds), "v"(off), "s"(A) : "memory");
        }
        if constexpr (BCONTIG) {
            // B tile: 64x64 f32 = 1024 chunks -> 4 wave-instructions
            #pragma unroll
            for (int i = 0; i < 4; ++i) {
                int chunk = i * THREADS + tid;
                int n  = chunk >> 4;
                int c4 = (chunk & 15) << 2;
                unsigned lds = (unsigned)(uintptr_t)(&Bsm2[buf][n * LDSW + c4]);
                unsigned off = (unsigned)((n * (int)ldbN + k0 + c4) * 4);
                asm volatile("global_load_async_to_lds_b128 %0, %1, %2"
                             :: "v"(lds), "v"(off), "s"(Bm) : "memory");
            }
        } else {
            // P@V path: V accessed column-wise (ldbK = D) -> scalar staging
            #pragma unroll
            for (int i = 0; i < (BN * BK) / THREADS; ++i) {
                int idx = i * THREADS + tid;
                int n = idx >> 6, c = idx & 63;
                Bsm2[buf][n * LDSW + c] =
                    Bm[(long long)n * ldbN + (long long)(k0 + c) * ldbK];
            }
        }
    };

    v8f acc[2][2] = {};

    const int nk = Kdim / BK;
    issue_stage(0, 0);

    for (int kt = 0; kt < nk; ++kt) {
        const int cur = kt & 1;
        if (kt + 1 < nk) {
            issue_stage((kt + 1) * BK, cur ^ 1);
            // next stage (<= 12 or 8 wave-instructions) may stay in flight;
            // in-order completion => current stage is done.
            if constexpr (BCONTIG)
                asm volatile("s_wait_asynccnt 0xc" ::: "memory");
            else
                asm volatile("s_wait_asynccnt 0x8" ::: "memory");
        } else {
            asm volatile("s_wait_asynccnt 0x0" ::: "memory");
        }
        __syncthreads();   // stage kt visible to all waves

        const float* Ab = &Asm2[cur][0];
        const float* Bb = &Bsm2[cur][0];

        #pragma unroll
        for (int ks = 0; ks < BK; ks += 32) {
            // fragments (ISA 7.12.2 layouts), f32->f16 packed in-register
            F16Frag af[2], bf[2];
            #pragma unroll
            for (int mi = 0; mi < 2; ++mi) {
                const float2* arow = reinterpret_cast<const float2*>(
                    &Ab[(wm + mi * 16 + ml) * LDSW]);
                #pragma unroll
                for (int v = 0; v < 8; ++v) {
                    // A 16x32: dword v holds K = 2v + (v>=4 ? 8 : 0) + 8*half
                    int p = (ks >> 1) + v + ((v >= 4) ? 4 : 0) + 4 * hf;
                    float2 t = arow[p];
                    af[mi].u[v] = pk16(t.x, t.y);
                }
            }
            #pragma unroll
            for (int ni = 0; ni < 2; ++ni) {
                const float2* brow = reinterpret_cast<const float2*>(
                    &Bb[(wn + ni * 16 + ml) * LDSW]);
                #pragma unroll
                for (int v = 0; v < 8; ++v) {
                    // B 32x16: dword v holds K = 2v + (v>=4 ? 8 : 0) + 8*half
                    float2 t = brow[(ks >> 1) + v + ((v >= 4) ? 4 : 0) + 4 * hf];
                    bf[ni].u[v] = pk16(t.x, t.y);
                }
            }
            #pragma unroll
            for (int mi = 0; mi < 2; ++mi)
                #pragma unroll
                for (int ni = 0; ni < 2; ++ni)
                    acc[mi][ni] = __builtin_amdgcn_wmma_f32_16x16x32_f16(
                        false, af[mi].h, false, bf[ni].h,
                        (short)0, acc[mi][ni], false, false);
        }
        __syncthreads();   // all reads of buffer `cur` done before it is refilled
    }

    // D/C layout: col = lane%16, row = r + 8*(lane/16)
    #pragma unroll
    for (int mi = 0; mi < 2; ++mi) {
        #pragma unroll
        for (int ni = 0; ni < 2; ++ni) {
            long long col = tileN + wn + ni * 16 + ml;
            float bval = bias ? bias[col] : 0.0f;
            #pragma unroll
            for (int r = 0; r < 8; ++r) {
                long long row = tileM + wm + mi * 16 + r + 8 * hf;
                float val = scale * acc[mi][ni][r] + bval;
                if constexpr (NT_C)
                    __builtin_nontemporal_store(val, &C[row * ldc + col]);
                else
                    C[row * ldc + col] = val;
            }
        }
    }
}

// In-place row softmax, float4-vectorized: one 256-thread block per row.
__global__ __launch_bounds__(THREADS)
void softmax_rows(float* __restrict__ scores, int ncols)
{
    __shared__ float red[THREADS];
    float4* p4 = reinterpret_cast<float4*>(scores + (long long)blockIdx.x * ncols);
    const int n4 = ncols >> 2;
    const int t = threadIdx.x;

    float lm = -3.402823466e38f;
    for (int i = t; i < n4; i += THREADS) {
        float4 x = p4[i];
        lm = fmaxf(lm, fmaxf(fmaxf(x.x, x.y), fmaxf(x.z, x.w)));
    }
    red[t] = lm; __syncthreads();
    for (int s = THREADS / 2; s > 0; s >>= 1) {
        if (t < s) red[t] = fmaxf(red[t], red[t + s]);
        __syncthreads();
    }
    const float m = red[0];
    __syncthreads();

    float ls = 0.0f;
    for (int i = t; i < n4; i += THREADS) {
        float4 x = p4[i];
        x.x = __expf(x.x - m); x.y = __expf(x.y - m);
        x.z = __expf(x.z - m); x.w = __expf(x.w - m);
        p4[i] = x;
        ls += x.x + x.y + x.z + x.w;
    }
    red[t] = ls; __syncthreads();
    for (int s = THREADS / 2; s > 0; s >>= 1) {
        if (t < s) red[t] += red[t + s];
        __syncthreads();
    }
    const float inv = 1.0f / red[0];
    for (int i = t; i < n4; i += THREADS) {
        float4 x = p4[i];
        x.x *= inv; x.y *= inv; x.z *= inv; x.w *= inv;
        p4[i] = x;
    }
}

extern "C" void kernel_launch(void* const* d_in, const int* in_sizes, int n_in,
                              void* d_out, int out_size, void* d_ws, size_t ws_size,
                              hipStream_t stream)
{
    const float* q  = (const float*)d_in[0];
    const float* k  = (const float*)d_in[1];
    const float* v  = (const float*)d_in[2];
    const float* Wq = (const float*)d_in[3];
    const float* bq = (const float*)d_in[4];
    const float* Wk = (const float*)d_in[5];
    const float* bk = (const float*)d_in[6];
    const float* Wv = (const float*)d_in[7];
    const float* bv = (const float*)d_in[8];
    const float* Wo = (const float*)d_in[9];
    const float* bo = (const float*)d_in[10];

    const int B = 2, S = 2048, D = 768, H = 12, DH = 64;
    const long long SD = (long long)S * D;
    const long long SS = (long long)S * S;

    float* out = (float*)d_out;                 // (B,S,D)
    float* P   = out + (long long)B * SD;       // scores region (B,H,S,S)

    float* Qb = (float*)d_ws;                   // 4 x 12.6 MB f32 scratch
    float* Kb = Qb + B * SD;
    float* Vb = Kb + B * SD;
    float* Cx = Vb + B * SD;

    dim3 blk(THREADS);
    const int Mrows = B * S; // 4096, batches contiguous in (B,S,D)

    // Q/K/V projections: X (4096x768) @ W^T (768x768) + bias
    wmma_gemm_f32<true, false, false><<<dim3(Mrows / BM, D / BN, 1), blk, 0, stream>>>(
        q, 0, 0, D,   Wq, 0, 0, D, 1,   Qb, 0, 0, D,   D, 1, bq, 1.0f);
    wmma_gemm_f32<true, false, false><<<dim3(Mrows / BM, D / BN, 1), blk, 0, stream>>>(
        k, 0, 0, D,   Wk, 0, 0, D, 1,   Kb, 0, 0, D,   D, 1, bk, 1.0f);
    wmma_gemm_f32<true, false, false><<<dim3(Mrows / BM, D / BN, 1), blk, 0, stream>>>(
        v, 0, 0, D,   Wv, 0, 0, D, 1,   Vb, 0, 0, D,   D, 1, bv, 1.0f);

    // Per-head logits: P = (Q_h @ K_h^T)/8 -> scores region; NT stores so the
    // 402MB stream does not evict the reused Q/K tiles from the 192MB L2.
    wmma_gemm_f32<true, false, true><<<dim3(S / BM, S / BN, B * H), blk, 0, stream>>>(
        Qb, SD, DH, D,
        Kb, SD, DH, D, 1,
        P, (long long)H * SS, SS, S,
        DH, H, nullptr, 0.125f);

    // Row softmax in place (B*H*S = 49152 rows of 2048)
    softmax_rows<<<dim3(B * H * S), blk, 0, stream>>>(P, S);

    // Per-head context: ctx_h = P_h @ V_h; P streamed NT (once-read 402MB),
    // V column-wise via scalar staging.
    wmma_gemm_f32<false, true, false><<<dim3(S / BM, DH / BN, B * H), blk, 0, stream>>>(
        P, (long long)H * SS, SS, S,
        Vb, SD, DH, 1, D,
        Cx, SD, DH, D,
        S, H, nullptr, 1.0f);

    // Output projection: out = ctx (4096x768) @ Wo^T + bo (ctx fits in L2, keep RT)
    wmma_gemm_f32<true, false, false><<<dim3(Mrows / BM, D / BN, 1), blk, 0, stream>>>(
        Cx, 0, 0, D,  Wo, 0, 0, D, 1,  out, 0, 0, D,  D, 1, bo, 1.0f);
}
